// Decoder_30468497998259
// MI455X (gfx1250) — matrix operand order repaired and checked by reading, not compiled, not run on hardware
//
#include <hip/hip_runtime.h>

#define BD   128      // batch
#define TT   128      // timesteps
#define HH   1024     // encoder hidden
#define HIDD 1024     // GRU hidden
#define VV   8000     // maxout units
#define NV2  16000    // V * P

typedef __attribute__((ext_vector_type(16))) __bf16        v16bf;
typedef __attribute__((ext_vector_type(8)))  __bf16        v8bf;
typedef __attribute__((ext_vector_type(8)))  float         v8f;
typedef __attribute__((ext_vector_type(8)))  unsigned short v8u16;

__device__ __forceinline__ unsigned short f2bf(float f) {
  unsigned u = __builtin_bit_cast(unsigned, f);
  u += 0x7FFFu + ((u >> 16) & 1u);              // round-to-nearest-even
  return (unsigned short)(u >> 16);
}
__device__ __forceinline__ float bf2f(unsigned short b) {
  return __builtin_bit_cast(float, ((unsigned)b) << 16);
}
__device__ __forceinline__ float sigmf(float x) { return 1.0f / (1.0f + expf(-x)); }

// Load 16 contiguous bf16 (32B) as a WMMA fragment half-pair.
__device__ __forceinline__ v16bf ld_frag16(const unsigned short* p) {
  const v8u16* q = (const v8u16*)p;
  v8bf x = __builtin_bit_cast(v8bf, q[0]);
  v8bf y = __builtin_bit_cast(v8bf, q[1]);
  v16bf r;
#pragma unroll
  for (int i = 0; i < 8; ++i) { r[i] = x[i]; r[8 + i] = y[i]; }
  return r;
}

// ---------------------------------------------------------------------------
// bf16 WMMA GEMM:  C[M,N] = act( A[M,K] * Bt[N,K]^T + bias + add )
//   A row-major bf16, Bt row-major bf16 ([N,K] = B^T), fp32 accumulate.
//   Block = 256 threads = 8 waves; block tile 16(M) x 512(N);
//   wave tile 16 x 64 = 4 WMMA tiles sharing one A fragment.
//   REQUIRES: M % 16 == 0, N % 64 == 0, K % 32 == 0 (all call sites comply).
//   A tile (16 x K bf16) staged in LDS, shared by all 8 waves.
// ---------------------------------------------------------------------------
__global__ __launch_bounds__(256)
void gemm_bf16_wmma(const unsigned short* __restrict__ A,
                    const unsigned short* __restrict__ Bt,
                    const float* __restrict__ bias,
                    const unsigned short* __restrict__ addBf, long long addStride,
                    float* __restrict__ Cf, unsigned short* __restrict__ Cbf,
                    int M, int N, int K, int act)
{
  extern __shared__ unsigned short As[];   // 16 * K bf16
  const int tid  = threadIdx.x;
  const int lane = tid & 31;
  const int wave = tid >> 5;
  const int m0   = blockIdx.y * 16;
  const int n0   = blockIdx.x * 512 + wave * 64;

  // Cooperative copy of the 16 x K A tile into LDS (rows are contiguous).
  {
    const v8u16* Ag  = (const v8u16*)(A + (size_t)m0 * K);
    v8u16*       As8 = (v8u16*)As;
    const int chunks = (16 * K) >> 3;
    for (int i = tid; i < chunks; i += 256) As8[i] = Ag[i];
  }
  __syncthreads();

  // Wave-uniform tail exit: N is a multiple of 64, so a wave's 64 columns
  // are either fully valid or fully out of range. No guards in the K-loop.
  if (n0 >= N) return;

  const int rhalf = lane >> 4;      // 0: lanes 0-15, 1: lanes 16-31
  const int l15   = lane & 15;

  v8f acc[4] = {{}, {}, {}, {}};
  const v8u16* Arow = (const v8u16*)(As + (size_t)l15 * K);
  const unsigned short* Bp[4];
#pragma unroll
  for (int tle = 0; tle < 4; ++tle)
    Bp[tle] = Bt + (size_t)(n0 + 16 * tle + l15) * K + 16 * rhalf;

#pragma unroll 2
  for (int k0 = 0; k0 < K; k0 += 32) {
    // A fragment: lane<16 -> K{k0..k0+15}; lane>=16 -> K{k0+16..k0+31}
    v16bf a;
    {
      v8u16 alo = Arow[(k0 >> 3) + 2 * rhalf];
      v8u16 ahi = Arow[(k0 >> 3) + 2 * rhalf + 1];
      v8bf x = __builtin_bit_cast(v8bf, alo);
      v8bf y = __builtin_bit_cast(v8bf, ahi);
#pragma unroll
      for (int i = 0; i < 8; ++i) { a[i] = x[i]; a[8 + i] = y[i]; }
    }
    if (k0 + 256 < K) {
      __builtin_prefetch(Bp[0] + k0 + 256, 0, 1);
      __builtin_prefetch(Bp[2] + k0 + 256, 0, 1);
    }
    // 4 B fragments (each lane: 16 contiguous K of its column), 4 WMMAs.
    v16bf b0 = ld_frag16(Bp[0] + k0);
    v16bf b1 = ld_frag16(Bp[1] + k0);
    v16bf b2 = ld_frag16(Bp[2] + k0);
    v16bf b3 = ld_frag16(Bp[3] + k0);
    acc[0] = __builtin_amdgcn_wmma_f32_16x16x32_bf16(false, a, false, b0,
                                                     (short)0, acc[0], false, false);
    acc[1] = __builtin_amdgcn_wmma_f32_16x16x32_bf16(false, a, false, b1,
                                                     (short)0, acc[1], false, false);
    acc[2] = __builtin_amdgcn_wmma_f32_16x16x32_bf16(false, a, false, b2,
                                                     (short)0, acc[2], false, false);
    acc[3] = __builtin_amdgcn_wmma_f32_16x16x32_bf16(false, a, false, b3,
                                                     (short)0, acc[3], false, false);
  }

  // Epilogue: C layout -> VGPR r of tile tle holds
  //   (m0 + r + 8*rhalf, n0 + 16*tle + lane&15). No bounds checks needed.
#pragma unroll
  for (int tle = 0; tle < 4; ++tle) {
    const int n = n0 + 16 * tle + l15;
#pragma unroll
    for (int r = 0; r < 8; ++r) {
      const int m = m0 + r + 8 * rhalf;
      float v = acc[tle][r];
      if (bias)  v += bias[n];
      if (addBf) v += bf2f(addBf[(size_t)m * addStride + n]);
      if (act)   v = tanhf(v);
      if (Cf)    Cf [(size_t)m * N + n] = v;
      if (Cbf)   Cbf[(size_t)m * N + n] = f2bf(v);
    }
  }
}

// ---------------------------------------------------------------------------
// Prep kernels
// ---------------------------------------------------------------------------
__global__ void convert_f32_bf16(const float* __restrict__ src,
                                 unsigned short* __restrict__ dst, long long n)
{
  long long i = (long long)blockIdx.x * blockDim.x + threadIdx.x;
  long long stride = (long long)gridDim.x * blockDim.x;
  for (; i < n; i += stride) dst[i] = f2bf(src[i]);
}

// dst[c*R + r] = bf16(src[(rowOff + r)*C + c])
__global__ void transpose_f32_bf16(const float* __restrict__ src,
                                   unsigned short* __restrict__ dst,
                                   int R, int C, int rowOff)
{
  int i = blockIdx.x * blockDim.x + threadIdx.x;
  if (i < R * C) {
    int r = i / C, c = i % C;
    dst[(size_t)c * R + r] = f2bf(src[(size_t)(rowOff + r) * C + c]);
  }
}

__global__ void init_state(float* sF, unsigned short* sB, int n)
{
  int i = blockIdx.x * blockDim.x + threadIdx.x;
  if (i < n) { sF[i] = 0.0f; sB[i] = 0; }
}

// ---------------------------------------------------------------------------
// Per-step VALU kernels
// ---------------------------------------------------------------------------
__global__ __launch_bounds__(128)
void softmax_T(const float* __restrict__ a, float* __restrict__ e)
{
  __shared__ float red[TT];
  const int b = blockIdx.x, tid = threadIdx.x;
  float v = a[(size_t)b * TT + tid];
  red[tid] = v; __syncthreads();
  for (int s = TT / 2; s > 0; s >>= 1) {
    if (tid < s) red[tid] = fmaxf(red[tid], red[tid + s]);
    __syncthreads();
  }
  float mx = red[0]; __syncthreads();
  float ex = expf(v - mx);
  red[tid] = ex; __syncthreads();
  for (int s = TT / 2; s > 0; s >>= 1) {
    if (tid < s) red[tid] += red[tid + s];
    __syncthreads();
  }
  e[(size_t)b * TT + tid] = ex / red[0];
}

// c[b,h] = sum_t e[b,t] * h[b,t,h]  (bf16 out for the GRU GEMM)
__global__ __launch_bounds__(256)
void context_kernel(const float* __restrict__ h, const float* __restrict__ e,
                    unsigned short* __restrict__ cB)
{
  __shared__ float es[TT];
  const int b  = blockIdx.x >> 2;
  const int hh = ((blockIdx.x & 3) << 8) + threadIdx.x;
  if (threadIdx.x < TT) es[threadIdx.x] = e[(size_t)b * TT + threadIdx.x];
  __syncthreads();
  const float* hb = h + (size_t)b * TT * HH + hh;
  float acc = 0.0f;
#pragma unroll 4
  for (int t = 0; t < TT; ++t) acc = fmaf(es[t], hb[(size_t)t * HH], acc);
  cB[(size_t)b * HH + hh] = f2bf(acc);
}

// PyTorch GRU cell gates; updates s (f32) and s_bf (bf16) in place.
__global__ __launch_bounds__(256)
void gru_gates(const float* __restrict__ gi, const float* __restrict__ gh,
               float* __restrict__ sF, unsigned short* __restrict__ sB)
{
  const int idx = blockIdx.x * blockDim.x + threadIdx.x;   // B*HID threads
  const int b = idx >> 10, j = idx & 1023;
  const size_t g = (size_t)b * 3 * HIDD + j;
  float r = sigmf(gi[g]          + gh[g]);
  float z = sigmf(gi[g + HIDD]   + gh[g + HIDD]);
  float n = tanhf(gi[g + 2*HIDD] + r * gh[g + 2*HIDD]);
  float sn = (1.0f - z) * n + z * sF[idx];
  sF[idx] = sn;
  sB[idx] = f2bf(sn);
}

// pairwise max over P=2 then softmax over V=8000, writing out[b, t, :]
__global__ __launch_bounds__(256)
void maxout_softmax(const float* __restrict__ m, float* __restrict__ out, int t)
{
  __shared__ float red[256];
  const int b = blockIdx.x, tid = threadIdx.x;
  const float* row_m = m + (size_t)b * NV2;
  float*       row_o = out + ((size_t)b * TT + t) * VV;

  float mx = -3.4e38f;
  for (int v = tid; v < VV; v += 256) {
    float pm = fmaxf(row_m[2 * v], row_m[2 * v + 1]);
    row_o[v] = pm;
    mx = fmaxf(mx, pm);
  }
  red[tid] = mx; __syncthreads();
  for (int s = 128; s > 0; s >>= 1) {
    if (tid < s) red[tid] = fmaxf(red[tid], red[tid + s]);
    __syncthreads();
  }
  mx = red[0]; __syncthreads();

  float sum = 0.0f;
  for (int v = tid; v < VV; v += 256) {
    float ex = expf(row_o[v] - mx);
    row_o[v] = ex;
    sum += ex;
  }
  red[tid] = sum; __syncthreads();
  for (int s = 128; s > 0; s >>= 1) {
    if (tid < s) red[tid] += red[tid + s];
    __syncthreads();
  }
  const float inv = 1.0f / red[0];
  for (int v = tid; v < VV; v += 256) row_o[v] *= inv;
}

// ---------------------------------------------------------------------------
extern "C" void kernel_launch(void* const* d_in, const int* in_sizes, int n_in,
                              void* d_out, int out_size, void* d_ws, size_t ws_size,
                              hipStream_t stream)
{
  (void)in_sizes; (void)n_in; (void)out_size; (void)ws_size;
  const float* hF  = (const float*)d_in[0];   // [B,T,H]
  const float* Wa1 = (const float*)d_in[1];   // [2048,1024]
  const float* ba1 = (const float*)d_in[2];   // [1024]
  const float* Wa2 = (const float*)d_in[3];   // [1024,128]
  const float* ba2 = (const float*)d_in[4];   // [128]
  const float* Wi  = (const float*)d_in[5];   // [3072,1024]
  const float* Wh  = (const float*)d_in[6];   // [3072,1024]
  const float* bi  = (const float*)d_in[7];   // [3072]
  const float* bh  = (const float*)d_in[8];   // [3072]
  const float* Wm  = (const float*)d_in[9];   // [16000,1024] flat
  const float* bm  = (const float*)d_in[10];  // [16000] flat
  float* out = (float*)d_out;                 // [B,T,V]

  // Workspace carve-out
  char* p = (char*)d_ws;
  auto carve = [&](size_t bytes) { char* r = p; p += (bytes + 255) & ~(size_t)255; return (void*)r; };
  unsigned short* Wa1Ts = (unsigned short*)carve((size_t)1024 * 1024 * 2); // s-half, [N=1024,K=1024]
  unsigned short* Wa1Th = (unsigned short*)carve((size_t)1024 * 1024 * 2); // h-half
  unsigned short* Wa2T  = (unsigned short*)carve((size_t)128  * 1024 * 2); // [N=128,K=1024]
  unsigned short* WiB   = (unsigned short*)carve((size_t)3072 * 1024 * 2);
  unsigned short* WhB   = (unsigned short*)carve((size_t)3072 * 1024 * 2);
  unsigned short* WmB   = (unsigned short*)carve((size_t)NV2  * 1024 * 2);
  unsigned short* hB    = (unsigned short*)carve((size_t)BD * TT * HH * 2);
  unsigned short* UhB   = (unsigned short*)carve((size_t)BD * TT * 1024 * 2); // h@Wa1_h, bf16
  unsigned short* X1    = (unsigned short*)carve((size_t)BD * 1024 * 2);      // tanh(...) bf16
  float*          aBuf  = (float*)carve((size_t)BD * TT * 4);
  float*          eBuf  = (float*)carve((size_t)BD * TT * 4);
  unsigned short* cB    = (unsigned short*)carve((size_t)BD * HH * 2);
  float*          giB   = (float*)carve((size_t)BD * 3072 * 4);
  float*          ghB   = (float*)carve((size_t)BD * 3072 * 4);
  float*          sF    = (float*)carve((size_t)BD * HIDD * 4);
  unsigned short* sB    = (unsigned short*)carve((size_t)BD * HIDD * 2);
  float*          mBuf  = (float*)carve((size_t)BD * NV2 * 4);

  const size_t ldsK = 16 * 1024 * sizeof(unsigned short); // 32 KB A tile

  // ---- one-time prep (re-run every launch; deterministic) ----
  convert_f32_bf16<<<2048, 256, 0, stream>>>(Wi, WiB, (long long)3072 * 1024);
  convert_f32_bf16<<<2048, 256, 0, stream>>>(Wh, WhB, (long long)3072 * 1024);
  convert_f32_bf16<<<2048, 256, 0, stream>>>(Wm, WmB, (long long)NV2 * 1024);
  convert_f32_bf16<<<2048, 256, 0, stream>>>(hF, hB,  (long long)BD * TT * HH);
  transpose_f32_bf16<<<4096, 256, 0, stream>>>(Wa1, Wa1Ts, 1024, 1024, 0);
  transpose_f32_bf16<<<4096, 256, 0, stream>>>(Wa1, Wa1Th, 1024, 1024, 1024);
  transpose_f32_bf16<<<512,  256, 0, stream>>>(Wa2, Wa2T, 1024, 128, 0);

  // Hoisted encoder projection: Uh[b*T+t, :] = h[b,t,:] @ Wa1_h   (bf16 out)
  gemm_bf16_wmma<<<dim3(2, 1024), 256, ldsK, stream>>>(
      hB, Wa1Th, nullptr, nullptr, 0, nullptr, UhB, BD * TT, 1024, 1024, 0);

  init_state<<<512, 256, 0, stream>>>(sF, sB, BD * HIDD);

  // ---- sequential scan over T steps ----
  for (int t = 0; t < TT; ++t) {
    // X1 = tanh(s @ Wa1_s + Uh[:,t,:] + ba1)        [128,1024] bf16
    gemm_bf16_wmma<<<dim3(2, 8), 256, ldsK, stream>>>(
        sB, Wa1Ts, ba1, UhB + (size_t)t * 1024, (long long)TT * 1024,
        nullptr, X1, BD, 1024, 1024, 1);
    // a = X1 @ Wa2 + ba2                            [128,128] f32
    gemm_bf16_wmma<<<dim3(1, 8), 256, ldsK, stream>>>(
        X1, Wa2T, ba2, nullptr, 0, aBuf, nullptr, BD, TT, 1024, 0);
    // e = softmax(a)
    softmax_T<<<BD, TT, 0, stream>>>(aBuf, eBuf);
    // c = einsum('bth,bt->bh', h, e)                bf16
    context_kernel<<<BD * 4, 256, 0, stream>>>(hF, eBuf, cB);
    // gi = c @ Wi^T + bi ; gh = s @ Wh^T + bh       [128,3072] f32
    gemm_bf16_wmma<<<dim3(6, 8), 256, ldsK, stream>>>(
        cB, WiB, bi, nullptr, 0, giB, nullptr, BD, 3072, 1024, 0);
    gemm_bf16_wmma<<<dim3(6, 8), 256, ldsK, stream>>>(
        sB, WhB, bh, nullptr, 0, ghB, nullptr, BD, 3072, 1024, 0);
    // GRU gate fusion -> s_new (f32 + bf16)
    gru_gates<<<512, 256, 0, stream>>>(giB, ghB, sF, sB);
    // m = s_new @ Wm^T + bm                         [128,16000] f32 (dominant GEMM)
    gemm_bf16_wmma<<<dim3(32, 8), 256, ldsK, stream>>>(
        sB, WmB, bm, nullptr, 0, mBuf, nullptr, BD, NV2, 1024, 0);
    // maxout pool + softmax over V -> out[:, t, :]
    maxout_softmax<<<BD, 256, 0, stream>>>(mBuf, out, t);
  }
}
